// ExpertLinear_78726750536044
// MI455X (gfx1250) — matrix operand order repaired and checked by reading, not compiled, not run on hardware
//
#include <hip/hip_runtime.h>

// ExpertLinear: y[b,o] = sum_k sum_i (ew[b,k]*x[b,i]) * W[k,o,i] + sum_k ew[b,k]*bias[k,o]
// B=512, E=8, IN=OUT=1024.
//
// Pass 1a: W (f32, 32MB) -> bf16 (16MB) once into d_ws.
// Pass 1b: xs[e][b][i] = bf16(ew[b,e] * x[b,i]) (8MB) into d_ws (folds the expert
//          blend into the A matrix, so the GEMM is 8 chained plain bf16 GEMMs).
// Pass 2:  blended GEMM. A and B tiles staged global->LDS with
//          global_load_async_to_lds_b128 (ASYNCcnt, triple-buffered, partial
//          s_wait_asynccnt for 2-deep prefetch); LDS rows padded to 80B for
//          bank-conflict-free ds_load_b128 fragment reads; math is
//          v_wmma_f32_16x16x32_bf16 with f32 accumulation.

#define BSZ  512
#define NEXP 8
#define DIN  1024
#define DOUT 1024
#define BM   64
#define BN   64
#define BK   32
#define LSTRIDE 80                           // 64B of bf16 + 16B pad -> conflict-free banks
#define NITER (NEXP * (DIN / BK))            // 256 K-steps total
#define WB_ELEMS ((size_t)NEXP * DOUT * DIN) // bf16 weight elements

typedef __attribute__((ext_vector_type(16))) __bf16       bf16x16;
typedef __attribute__((ext_vector_type(8)))  float        f32x8;
typedef __attribute__((ext_vector_type(4)))  float        f32x4;
typedef __attribute__((ext_vector_type(4)))  unsigned int u32x4;

// Exact parameter type of the async-copy builtin (from compiler diagnostics):
// pointer to gcc-style int4 vector, AS1 (global) / AS3 (LDS).
typedef int v4i_vs __attribute__((vector_size(16)));
#define AS1P __attribute__((address_space(1))) v4i_vs*
#define AS3P __attribute__((address_space(3))) v4i_vs*

#if defined(__has_builtin)
#if __has_builtin(__builtin_amdgcn_global_load_async_to_lds_b128)
#define HAVE_ASYNC_B128 1
#endif
#if __has_builtin(__builtin_amdgcn_s_wait_asynccnt)
#define HAVE_WAIT_ASYNC 1
#endif
#endif

union FragCast { u32x4 q[2]; bf16x16 f; };

static __device__ __forceinline__ unsigned int pack_bf16(float a, float b) {
  union { __bf16 h[2]; unsigned int u; } p;
  p.h[0] = (__bf16)a;
  p.h[1] = (__bf16)b;
  return p.u;   // v_cvt_pk_bf16_f32
}

// ---------------- Pass 1a: W f32 -> bf16 ----------------
__global__ __launch_bounds__(256) void ConvertW_kernel(const float* __restrict__ w,
                                                       unsigned int* __restrict__ wb) {
  const size_t idx = ((size_t)blockIdx.x * 256 + threadIdx.x) * 8;  // elements
  f32x4 v0 = *(const f32x4*)(w + idx);
  f32x4 v1 = *(const f32x4*)(w + idx + 4);
  u32x4 d;
  d.x = pack_bf16(v0.x, v0.y);
  d.y = pack_bf16(v0.z, v0.w);
  d.z = pack_bf16(v1.x, v1.y);
  d.w = pack_bf16(v1.z, v1.w);
  *(u32x4*)(wb + idx / 2) = d;
}

// ---------------- Pass 1b: xs[e][b][i] = bf16(ew[b,e]*x[b,i]) ----------------
__global__ __launch_bounds__(256) void ScaleX_kernel(const float* __restrict__ x,
                                                     const float* __restrict__ ew,
                                                     unsigned int* __restrict__ xs) {
  const int t = blockIdx.x * 256 + threadIdx.x;  // 0 .. B*DIN/8-1
  const int b = t >> 7;                          // DIN/8 = 128 chunks per row
  const int c = (t & 127) * 8;
  f32x4 v0 = *(const f32x4*)(x + (size_t)b * DIN + c);
  f32x4 v1 = *(const f32x4*)(x + (size_t)b * DIN + c + 4);
  f32x4 e0 = *(const f32x4*)(ew + (size_t)b * NEXP);
  f32x4 e1 = *(const f32x4*)(ew + (size_t)b * NEXP + 4);
  const float es[NEXP] = {e0.x, e0.y, e0.z, e0.w, e1.x, e1.y, e1.z, e1.w};
#pragma unroll
  for (int e = 0; e < NEXP; ++e) {
    const float s = es[e];
    u32x4 d;
    d.x = pack_bf16(v0.x * s, v0.y * s);
    d.y = pack_bf16(v0.z * s, v0.w * s);
    d.z = pack_bf16(v1.x * s, v1.y * s);
    d.w = pack_bf16(v1.z * s, v1.w * s);
    *(u32x4*)(xs + ((size_t)e * BSZ + b) * (DIN / 2) + c / 2) = d;
  }
}

// ---------------- Pass 2: blended GEMM ----------------

// Copy 32B of bf16 data straight into LDS (per-lane, ASYNCcnt-tracked).
static __device__ __forceinline__ void issueTile(const unsigned short* __restrict__ g,
                                                 unsigned char* l) {
#if defined(HAVE_ASYNC_B128)
  __builtin_amdgcn_global_load_async_to_lds_b128((AS1P)g, (AS3P)l, 0,  0);
  __builtin_amdgcn_global_load_async_to_lds_b128((AS1P)g, (AS3P)l, 16, 0);
#else
  u32x4 d0 = ((const u32x4*)g)[0];
  u32x4 d1 = ((const u32x4*)g)[1];
  *(u32x4*)l        = d0;
  *(u32x4*)(l + 16) = d1;
#endif
}

template <int N>
static __device__ __forceinline__ void wait_async_le() {
#if defined(HAVE_ASYNC_B128)
#if defined(HAVE_WAIT_ASYNC)
  __builtin_amdgcn_s_wait_asynccnt(N);
#else
  asm volatile("s_wait_asynccnt %0" ::"i"(N) : "memory");
#endif
#endif
}

// A-fragment (16x32 bf16): lanes 0-15 -> K chunks [0..7],[16..23]; lanes 16-31 -> [8..15],[24..31]
static __device__ __forceinline__ bf16x16 ldsA_frag(const unsigned char* base, int row, int lh) {
  FragCast c;
  c.q[0] = *(const u32x4*)(base + row * LSTRIDE + lh * 16);
  c.q[1] = *(const u32x4*)(base + row * LSTRIDE + lh * 16 + 32);
  return c.f;
}

// B-fragment (32x16 bf16): lanes 0-15 -> K chunks [0..7],[16..23]; lanes 16-31 -> [8..15],[24..31]
static __device__ __forceinline__ bf16x16 ldsB_frag(const unsigned char* base, int row, int lh) {
  FragCast c;
  c.q[0] = *(const u32x4*)(base + row * LSTRIDE + lh * 16);
  c.q[1] = *(const u32x4*)(base + row * LSTRIDE + lh * 16 + 32);
  return c.f;
}

static __device__ __forceinline__ f32x8 wmma_bf16(bf16x16 a, bf16x16 b, f32x8 c) {
  return __builtin_amdgcn_wmma_f32_16x16x32_bf16(false, a, false, b, (short)0, c, false, false);
}

__global__ __launch_bounds__(128) void ExpertLinear_wmma_kernel(
    const unsigned short* __restrict__ xsb,  // [E, B, IN] bf16 (ew-scaled x)
    const float* __restrict__ ew,            // [B, E]  f32 (for bias blend)
    const unsigned short* __restrict__ wb,   // [E, OUT, IN] bf16
    const float* __restrict__ bias,          // [E, OUT] f32
    float* __restrict__ out)                 // [B, OUT] f32
{
  __shared__ __align__(16) unsigned char lA[3][BM * LSTRIDE];  // triple buffered
  __shared__ __align__(16) unsigned char lB[3][BN * LSTRIDE];
  __shared__ float ewS[BM * NEXP];
  __shared__ float bS[NEXP * BN];

  const int tid   = threadIdx.x;
  const int tileN = blockIdx.x * BN;
  const int tileM = blockIdx.y * BM;

  // Stage ew (64x8) and bias (8x64) once (epilogue only).
  {
    const int i4 = tid * 4;
    *(f32x4*)&ewS[i4] = *(const f32x4*)&ew[(size_t)tileM * NEXP + i4];
    const int e = i4 >> 6;
    const int j = i4 & 63;
    *(f32x4*)&bS[i4] = *(const f32x4*)&bias[(size_t)e * DOUT + tileN + j];
  }

  // Staging role: each thread copies half a row (32B) of each tile.
  const int row     = tid >> 1;
  const int halfsel = tid & 1;

  // 4 wave32s in 2x2; each computes a 32x32 macro-tile = 2x2 WMMA subtiles.
  const int wave = tid >> 5;
  const int lane = tid & 31;
  const int lm   = lane & 15;
  const int lh   = lane >> 4;
  const int wM   = (wave >> 1) * 32;
  const int wN   = (wave & 1) * 32;

  f32x8 zero = {0.f, 0.f, 0.f, 0.f, 0.f, 0.f, 0.f, 0.f};
  f32x8 acc00 = zero, acc01 = zero, acc10 = zero, acc11 = zero;

  // Issue one K-step's A+B tiles: 4 async b128 copies per thread.
  auto issueStep = [&](int ek, int buf) {
    const int e  = ek >> 5;
    const int k0 = (ek & 31) * BK + halfsel * 16;  // bf16 elements
    const unsigned short* ga = xsb + ((size_t)e * BSZ  + tileM + row) * DIN + k0;
    const unsigned short* gb = wb  + ((size_t)e * DOUT + tileN + row) * DIN + k0;
    unsigned char* la = lA[buf] + row * LSTRIDE + halfsel * 32;
    unsigned char* lb = lB[buf] + row * LSTRIDE + halfsel * 32;
    issueTile(ga, la);
    issueTile(gb, lb);
  };

  issueStep(0, 0);
  issueStep(1, 1);

  int cur = 0;
  for (int ek = 0; ek < NITER; ++ek) {
    // Step ek's 4 copies are the oldest outstanding; async loads complete in
    // order, so waiting "<= 4" retires them while step ek+1's stay in flight.
    if (ek + 1 < NITER) wait_async_le<4>();
    else                wait_async_le<0>();
    __syncthreads();
    if (ek + 2 < NITER) {
      int nb = cur + 2; if (nb >= 3) nb -= 3;
      issueStep(ek + 2, nb);
    }

    bf16x16 a0 = ldsA_frag(lA[cur], wM + 0  + lm, lh);
    bf16x16 a1 = ldsA_frag(lA[cur], wM + 16 + lm, lh);
    bf16x16 b0 = ldsB_frag(lB[cur], wN + 0  + lm, lh);
    bf16x16 b1 = ldsB_frag(lB[cur], wN + 16 + lm, lh);

    acc00 = wmma_bf16(a0, b0, acc00);
    acc01 = wmma_bf16(a0, b1, acc01);
    acc10 = wmma_bf16(a1, b0, acc10);
    acc11 = wmma_bf16(a1, b1, acc11);

    if (++cur == 3) cur = 0;
  }

  // Epilogue: blended bias + store. C layout: VGPR r -> M = r + 8*(lane/16), N = lane%16.
  f32x8 accs[2][2] = {{acc00, acc01}, {acc10, acc11}};
#pragma unroll
  for (int mt = 0; mt < 2; ++mt)
#pragma unroll
    for (int nt = 0; nt < 2; ++nt)
#pragma unroll
      for (int r = 0; r < 8; ++r) {
        const int Ml = wM + mt * 16 + r + 8 * lh;
        const int Nl = wN + nt * 16 + lm;
        float v = accs[mt][nt][r];
#pragma unroll
        for (int e = 0; e < NEXP; ++e) v += ewS[Ml * NEXP + e] * bS[e * BN + Nl];
        out[(size_t)(tileM + Ml) * DOUT + tileN + Nl] = v;
      }
}

extern "C" void kernel_launch(void* const* d_in, const int* in_sizes, int n_in,
                              void* d_out, int out_size, void* d_ws, size_t ws_size,
                              hipStream_t stream) {
  (void)in_sizes; (void)n_in; (void)out_size; (void)ws_size;
  const float* x    = (const float*)d_in[0];  // [512,1024]
  const float* ew   = (const float*)d_in[1];  // [512,8]
  const float* w    = (const float*)d_in[2];  // [8,1024,1024]
  const float* bias = (const float*)d_in[3];  // [8,1024]
  float* out        = (float*)d_out;          // [512,1024]

  unsigned int* wb = (unsigned int*)d_ws;                          // 16 MB bf16 weights
  unsigned int* xs = (unsigned int*)((char*)d_ws + WB_ELEMS * 2);  // 8 MB bf16 scaled x

  ConvertW_kernel<<<dim3((NEXP * DOUT * DIN) / (256 * 8)), dim3(256), 0, stream>>>(w, wb);
  ScaleX_kernel<<<dim3((BSZ * DIN / 8) / 256), dim3(256), 0, stream>>>(x, ew, xs);
  ExpertLinear_wmma_kernel<<<dim3(DOUT / BN, BSZ / BM), dim3(128), 0, stream>>>(
      (const unsigned short*)xs, ew, (const unsigned short*)wb, bias, out);
}
